// SparseFFFKernel_15908558864463
// MI455X (gfx1250) — matrix operand order repaired and checked by reading, not compiled, not run on hardware
//
#include <hip/hip_runtime.h>
#include <math.h>

// ---- problem constants (from reference) ----
#define HIDDEN 4096
#define INTER  16384
#define KACT   655      // int(16384 * 0.04)
#define KP     704      // KACT padded to 64 (11 wave-N chunks of 64)
#define TOKENS 16384    // BATCH*SEQ = 4*4096

typedef __attribute__((ext_vector_type(16))) __bf16          v16bf;
typedef __attribute__((ext_vector_type(8)))  float           v8f;
typedef __attribute__((ext_vector_type(16))) unsigned short  v16u;
typedef __attribute__((ext_vector_type(8)))  unsigned short  v8u;

// ---- f32 -> bf16 round-to-nearest-even (inputs are finite; no inf/nan guard) ----
__device__ __forceinline__ unsigned short f32_to_bf16(float f) {
    unsigned int u = __float_as_uint(f);
    unsigned int r = u + 0x7FFFu + ((u >> 16) & 1u);
    return (unsigned short)(r >> 16);
}

// Load a 16x32 A-fragment half-pair: lane holds K in {kh*16..kh*16+15}
__device__ __forceinline__ v16bf load_a_frag(const unsigned short* arow, int k0) {
    const v8u lo = *(const v8u*)(arow + k0);         // 16B
    const v8u hi = *(const v8u*)(arow + k0 + 8);     // 16B
    const v16u u = __builtin_shufflevector(lo, hi, 0, 1, 2, 3, 4, 5, 6, 7,
                                                   8, 9, 10, 11, 12, 13, 14, 15);
    return __builtin_bit_cast(v16bf, u);
}

// =====================================================================
// Kernel 0: one-time x -> bf16 conversion (contiguous, vectorized)
// =====================================================================
__global__ __launch_bounds__(256) void k_convert_x(
    const float* __restrict__ x, unsigned short* __restrict__ xb)
{
    const size_t base = ((size_t)blockIdx.x * 256 + threadIdx.x) * 16;
    const float4* src = (const float4*)(x + base);
    v16u u;
    #pragma unroll
    for (int q = 0; q < 4; ++q) {
        const float4 f = src[q];
        u[q * 4 + 0] = f32_to_bf16(f.x);
        u[q * 4 + 1] = f32_to_bf16(f.y);
        u[q * 4 + 2] = f32_to_bf16(f.z);
        u[q * 4 + 3] = f32_to_bf16(f.w);
    }
    *(v16u*)(xb + base) = u;    // 32B store
}

// =====================================================================
// Kernel 1: rowAbs[i] = sum_h |W_up[i][h]|       (one block per row)
// =====================================================================
__global__ __launch_bounds__(256) void k_row_abs_sum(
    const float* __restrict__ Wup, float* __restrict__ rowAbs)
{
    __shared__ float s[256];
    const int row = blockIdx.x;
    const float* p = Wup + (size_t)row * HIDDEN;
    float acc = 0.f;
    for (int c = threadIdx.x; c < HIDDEN; c += 256) acc += fabsf(p[c]);
    s[threadIdx.x] = acc;
    __syncthreads();
    for (int off = 128; off > 0; off >>= 1) {
        if (threadIdx.x < off) s[threadIdx.x] += s[threadIdx.x + off];
        __syncthreads();
    }
    if (threadIdx.x == 0) rowAbs[row] = s[0];
}

// =====================================================================
// Kernel 2: imp[j] = rowAbs[j] * sum_r |W_down[r][j]|   (coalesced cols)
// =====================================================================
__global__ __launch_bounds__(256) void k_col_importance(
    const float* __restrict__ Wdown, const float* __restrict__ rowAbs,
    float* __restrict__ imp)
{
    const int j = blockIdx.x * 256 + threadIdx.x;
    float acc = 0.f;
    #pragma unroll 4
    for (int r = 0; r < HIDDEN; ++r) acc += fabsf(Wdown[(size_t)r * INTER + j]);
    imp[j] = rowAbs[j] * acc;
}

// =====================================================================
// Kernel 3: top-K select (single block, 1024 threads).
// Keys are non-negative floats -> raw uint bits are order-preserving.
// Bitwise threshold search + deterministic scan-based compaction.
// =====================================================================
__global__ __launch_bounds__(1024) void k_topk(
    const float* __restrict__ imp, int* __restrict__ idxOut)
{
    __shared__ unsigned s[1024];
    const int t = threadIdx.x;

    unsigned key[16];
    #pragma unroll
    for (int i = 0; i < 16; ++i) {
        float v = imp[i * 1024 + t];
        key[i] = __float_as_uint(v < 0.f ? 0.f : v);
    }

    // Find threshold = K-th largest key (bitwise descent)
    unsigned thr = 0u;
    for (int bit = 31; bit >= 0; --bit) {
        const unsigned cand = thr | (1u << bit);
        unsigned local = 0;
        #pragma unroll
        for (int i = 0; i < 16; ++i) local += (key[i] >= cand);
        s[t] = local;
        __syncthreads();
        for (int off = 512; off > 0; off >>= 1) {
            if (t < off) s[t] += s[t + off];
            __syncthreads();
        }
        const unsigned cnt = s[0];
        __syncthreads();
        if (cnt >= (unsigned)KACT) thr = cand;
    }

    // Compact: all keys > thr, then first (K - count_gt) keys == thr.
    unsigned gcnt = 0, ecnt = 0;
    #pragma unroll
    for (int i = 0; i < 16; ++i) { gcnt += (key[i] > thr); ecnt += (key[i] == thr); }

    s[t] = gcnt;
    __syncthreads();
    for (int off = 1; off < 1024; off <<= 1) {
        unsigned v = (t >= off) ? s[t - off] : 0u;
        __syncthreads();
        s[t] += v;
        __syncthreads();
    }
    const unsigned totalGt = s[1023];
    unsigned pos = s[t] - gcnt;
    #pragma unroll
    for (int i = 0; i < 16; ++i)
        if (key[i] > thr) idxOut[pos++] = i * 1024 + t;
    __syncthreads();

    s[t] = ecnt;
    __syncthreads();
    for (int off = 1; off < 1024; off <<= 1) {
        unsigned v = (t >= off) ? s[t - off] : 0u;
        __syncthreads();
        s[t] += v;
        __syncthreads();
    }
    unsigned pos2 = totalGt + (s[t] - ecnt);
    #pragma unroll
    for (int i = 0; i < 16; ++i)
        if (key[i] == thr) { if (pos2 < (unsigned)KACT) idxOut[pos2] = i * 1024 + t; ++pos2; }
}

// =====================================================================
// Kernel 4: gather + convert up-weights: Wg[j][h] = bf16(W_up[idx[j]][h])
// =====================================================================
__global__ __launch_bounds__(256) void k_gather_up(
    const float* __restrict__ Wup, const int* __restrict__ idx,
    unsigned short* __restrict__ Wg)
{
    const int j = blockIdx.x;            // 0..KP-1
    unsigned short* dst = Wg + (size_t)j * HIDDEN;
    if (j < KACT) {
        const float* src = Wup + (size_t)idx[j] * HIDDEN;
        for (int h = threadIdx.x; h < HIDDEN; h += 256) dst[h] = f32_to_bf16(src[h]);
    } else {
        for (int h = threadIdx.x; h < HIDDEN; h += 256) dst[h] = 0;
    }
}

// =====================================================================
// Kernel 5: gather down-weights transposed: Dg[hh][j] = bf16(W_down[hh][idx[j]])
// =====================================================================
__global__ __launch_bounds__(256) void k_gather_down(
    const float* __restrict__ Wdown, const int* __restrict__ idx,
    unsigned short* __restrict__ Dg)
{
    const int hh = blockIdx.x;           // 0..HIDDEN-1
    const float* src = Wdown + (size_t)hh * INTER;
    unsigned short* dst = Dg + (size_t)hh * KP;
    for (int j = threadIdx.x; j < KP; j += 256)
        dst[j] = (j < KACT) ? f32_to_bf16(src[idx[j]]) : (unsigned short)0;
}

// =====================================================================
// GEMM cores.  Block = 128 threads (4 waves stacked in M); wave tile =
// 16M x 64N (4 f32 accumulators).  WMMA bf16 16x16x32, f32 accumulate.
// Grid: blockIdx.x = N-tile (fast -> A stripe stays in L2, small B matrix
// L2-resident), blockIdx.y = M-tile.
// Manual double-buffering: next iteration's A + 4 B fragments are loaded
// before the current 4 WMMAs so s_wait has slack.
// Fragment layouts used here (A/B contiguous k order, not the ISA table order):
//   A: lanes 0-15 row=lane, K=0..15 contiguous; lanes 16-31 K=16..31
//   B: lanes 0-15 col=lane, K=0..15 contiguous; lanes 16-31 K=16..31
//   D: lanes 0-15 N=lane, M=v; lanes 16-31 N=lane-16, M=v+8
// =====================================================================

#define WMMA_BF16(A, B, C) \
    __builtin_amdgcn_wmma_f32_16x16x32_bf16(false, (A), false, (B), (short)0, (C), false, false)

// Kernel 6: GEMM1 + exact GeLU:  Hg[t][j] = gelu(sum_h xb[t][h]*Wg[j][h])
__global__ __launch_bounds__(128) void k_gemm1_gelu(
    const unsigned short* __restrict__ xb, const unsigned short* __restrict__ Wg,
    unsigned short* __restrict__ Hg)
{
    const int lane = threadIdx.x & 31;
    const int wave = threadIdx.x >> 5;
    const int kh   = lane >> 4;
    const int l16  = lane & 15;
    const int n0 = blockIdx.x * 64;                  // fast dim: N (11 tiles)
    const int m0 = blockIdx.y * 64 + wave * 16;      // slow dim: M

    v8f c0 = {}, c1 = {}, c2 = {}, c3 = {};

    const unsigned short* arow = xb + (size_t)(m0 + l16) * HIDDEN + kh * 16;
    const unsigned short* br0  = Wg + (size_t)(n0 + l16) * HIDDEN + kh * 16;
    const unsigned short* br1  = br0 + (size_t)16 * HIDDEN;
    const unsigned short* br2  = br0 + (size_t)32 * HIDDEN;
    const unsigned short* br3  = br0 + (size_t)48 * HIDDEN;

    v16bf a  = load_a_frag(arow, 0);
    v16bf b0 = *(const v16bf*)(br0);
    v16bf b1 = *(const v16bf*)(br1);
    v16bf b2 = *(const v16bf*)(br2);
    v16bf b3 = *(const v16bf*)(br3);

    for (int k0 = 32; k0 < HIDDEN; k0 += 32) {
        const v16bf an  = load_a_frag(arow, k0);
        const v16bf b0n = *(const v16bf*)(br0 + k0);
        const v16bf b1n = *(const v16bf*)(br1 + k0);
        const v16bf b2n = *(const v16bf*)(br2 + k0);
        const v16bf b3n = *(const v16bf*)(br3 + k0);
        c0 = WMMA_BF16(a, b0, c0);
        c1 = WMMA_BF16(a, b1, c1);
        c2 = WMMA_BF16(a, b2, c2);
        c3 = WMMA_BF16(a, b3, c3);
        a = an; b0 = b0n; b1 = b1n; b2 = b2n; b3 = b3n;
    }
    c0 = WMMA_BF16(a, b0, c0);
    c1 = WMMA_BF16(a, b1, c1);
    c2 = WMMA_BF16(a, b2, c2);
    c3 = WMMA_BF16(a, b3, c3);

    v8f acc[4] = { c0, c1, c2, c3 };
    #pragma unroll
    for (int nt = 0; nt < 4; ++nt) {
        const int col = n0 + nt * 16 + l16;
        #pragma unroll
        for (int v = 0; v < 8; ++v) {
            const int row = m0 + v + kh * 8;
            float val = acc[nt][v];
            val = 0.5f * val * (1.0f + erff(val * 0.70710678118654752f));
            Hg[(size_t)row * KP + col] = f32_to_bf16(val);
        }
    }
}

// Kernel 7: GEMM2:  out[t][hh] = sum_j Hg[t][j] * Dg[hh][j]   (f32 out)
__global__ __launch_bounds__(128) void k_gemm2(
    const unsigned short* __restrict__ Hg, const unsigned short* __restrict__ Dg,
    float* __restrict__ out)
{
    const int lane = threadIdx.x & 31;
    const int wave = threadIdx.x >> 5;
    const int kh   = lane >> 4;
    const int l16  = lane & 15;
    const int n0 = blockIdx.x * 64;                  // fast dim: N (64 tiles)
    const int m0 = blockIdx.y * 64 + wave * 16;

    v8f c0 = {}, c1 = {}, c2 = {}, c3 = {};

    const unsigned short* arow = Hg + (size_t)(m0 + l16) * KP + kh * 16;
    const unsigned short* br0  = Dg + (size_t)(n0 + l16) * KP + kh * 16;
    const unsigned short* br1  = br0 + (size_t)16 * KP;
    const unsigned short* br2  = br0 + (size_t)32 * KP;
    const unsigned short* br3  = br0 + (size_t)48 * KP;

    v16bf a  = load_a_frag(arow, 0);
    v16bf b0 = *(const v16bf*)(br0);
    v16bf b1 = *(const v16bf*)(br1);
    v16bf b2 = *(const v16bf*)(br2);
    v16bf b3 = *(const v16bf*)(br3);

    for (int k0 = 32; k0 < KP; k0 += 32) {
        const v16bf an  = load_a_frag(arow, k0);
        const v16bf b0n = *(const v16bf*)(br0 + k0);
        const v16bf b1n = *(const v16bf*)(br1 + k0);
        const v16bf b2n = *(const v16bf*)(br2 + k0);
        const v16bf b3n = *(const v16bf*)(br3 + k0);
        c0 = WMMA_BF16(a, b0, c0);
        c1 = WMMA_BF16(a, b1, c1);
        c2 = WMMA_BF16(a, b2, c2);
        c3 = WMMA_BF16(a, b3, c3);
        a = an; b0 = b0n; b1 = b1n; b2 = b2n; b3 = b3n;
    }
    c0 = WMMA_BF16(a, b0, c0);
    c1 = WMMA_BF16(a, b1, c1);
    c2 = WMMA_BF16(a, b2, c2);
    c3 = WMMA_BF16(a, b3, c3);

    v8f acc[4] = { c0, c1, c2, c3 };
    #pragma unroll
    for (int nt = 0; nt < 4; ++nt) {
        const int col = n0 + nt * 16 + l16;
        #pragma unroll
        for (int v = 0; v < 8; ++v) {
            const int row = m0 + v + kh * 8;
            out[(size_t)row * HIDDEN + col] = acc[nt][v];
        }
    }
}

// =====================================================================
// Launch.  Workspace layout (bytes), total ~169 MB:
//   [0,       64K)   rowAbs   f32[16384]
//   [64K,    128K)   imp      f32[16384]
//   [128K,   132K)   idx      i32[KP]
//   [132K)           Wg       bf16[KP][4096]        (5.77 MB)
//   (+...)           Dg       bf16[4096][KP]        (5.77 MB)
//   (+...)           Hg       bf16[16384][KP]       (23.1 MB)
//   (+...)           xb       bf16[16384][4096]     (134.2 MB)
// =====================================================================
extern "C" void kernel_launch(void* const* d_in, const int* in_sizes, int n_in,
                              void* d_out, int out_size, void* d_ws, size_t ws_size,
                              hipStream_t stream) {
    const float* x     = (const float*)d_in[0];   // [4,4096,4096]
    const float* Wup   = (const float*)d_in[1];   // [16384,4096]
    const float* Wdown = (const float*)d_in[2];   // [4096,16384]
    float* out = (float*)d_out;                   // [4,4096,4096]

    char* ws = (char*)d_ws;
    float*          rowAbs = (float*)(ws);
    float*          imp    = (float*)(ws + (size_t)64 * 1024);
    int*            idx    = (int*)  (ws + (size_t)128 * 1024);
    unsigned short* Wg     = (unsigned short*)(ws + (size_t)132 * 1024);
    unsigned short* Dg     = Wg + (size_t)KP * HIDDEN;
    unsigned short* Hg     = Dg + (size_t)HIDDEN * KP;
    unsigned short* xb     = Hg + (size_t)TOKENS * KP;

    // x -> bf16 (independent of top-k path)
    k_convert_x     <<<(TOKENS * (HIDDEN / 16)) / 256, 256, 0, stream>>>(x, xb);

    k_row_abs_sum   <<<INTER, 256, 0, stream>>>(Wup, rowAbs);
    k_col_importance<<<INTER / 256, 256, 0, stream>>>(Wdown, rowAbs, imp);
    k_topk          <<<1, 1024, 0, stream>>>(imp, idx);
    k_gather_up     <<<KP, 256, 0, stream>>>(Wup, idx, Wg);
    k_gather_down   <<<HIDDEN, 256, 0, stream>>>(Wdown, idx, Dg);

    dim3 g1(KP / 64, TOKENS / 64);        // N fast (11), M slow (256)
    k_gemm1_gelu    <<<g1, 128, 0, stream>>>(xb, Wg, Hg);

    dim3 g2(HIDDEN / 64, TOKENS / 64);    // N fast (64), M slow (256)
    k_gemm2         <<<g2, 128, 0, stream>>>(Hg, Dg, out);
}
